// CausalAttention_50586124812645
// MI455X (gfx1250) — matrix operand order repaired and checked by reading, not compiled, not run on hardware
//
#include <hip/hip_runtime.h>

typedef __attribute__((ext_vector_type(16))) _Float16 v16h;
typedef __attribute__((ext_vector_type(8)))  _Float16 v8h;
typedef __attribute__((ext_vector_type(8)))  float    v8f;

#define EMBED 2048
#define NHEAD 16
#define HDIM  128
#define BATCH 2
#define SEQ   2048
#define KSTEPS (EMBED / 32)                           // 64
#define BHTD  ((size_t)BATCH * NHEAD * SEQ * HDIM)    // 8388608 elements

// ---------------------------------------------------------------------------
// CDNA5 WMMA helpers (wave32). Layouts per CDNA5 ISA §7.12.2:
//  A (16x32 f16): lane L holds row M=L&15; K in {kb..kb+7} U {kb+16..kb+23},
//                 kb = 8*(L>>4)
//  B (32x16 f16): lane L holds col N=L&15; K same as A (kb..kb+7, kb+16..kb+23)
//  C/D (16x16 f32, 8 VGPRs): element j -> row = j + 8*(L>>4), col = L&15
// ---------------------------------------------------------------------------
__device__ __forceinline__ v8f wmma16(v16h a, v16h b, v8f c) {
  return __builtin_amdgcn_wmma_f32_16x16x32_f16(false, a, false, b,
                                                (short)0, c, false, false);
}

__device__ __forceinline__ v16h load_a_frag(const _Float16* base, int ld) {
  const int lane = (int)(threadIdx.x & 31u);
  const _Float16* p = base + (lane & 15) * ld + ((lane >> 4) << 3);
  v16h a;
  ((v8h*)&a)[0] = *(const v8h*)(p);
  ((v8h*)&a)[1] = *(const v8h*)(p + 16);
  return a;
}

__device__ __forceinline__ v16h load_b_frag(const _Float16* base, int ld) {
  const int lane = (int)(threadIdx.x & 31u);
  const _Float16* p = base + (lane & 15) * ld + ((lane >> 4) << 3);
  v16h b;
  ((v8h*)&b)[0] = *(const v8h*)(p);
  ((v8h*)&b)[1] = *(const v8h*)(p + 16);
  return b;
}

// ---------------------------------------------------------------------------
// fp32-input GEMM tile staging (128x32 A tile, 32x128 B tile -> transposed)
// ---------------------------------------------------------------------------
struct TileRegsF32 {
  float4 a[4];
  float4 b[4];
};

__device__ __forceinline__ void stage_load_f32(
    const float* __restrict__ x, const float* __restrict__ W, int rowBase,
    int colBase, int k0, int xr, int xc4, int wk, int wn, TileRegsF32& t) {
#pragma unroll
  for (int rr = 0; rr < 4; ++rr)
    t.a[rr] = *(const float4*)(x + (size_t)(rowBase + xr + rr * 32) * EMBED +
                               k0 + xc4);
  const float* wp = W + (size_t)(k0 + wk) * EMBED + colBase + wn;
#pragma unroll
  for (int q4 = 0; q4 < 4; ++q4) t.b[q4] = *(const float4*)(wp + q4 * 4);
}

__device__ __forceinline__ void stage_store_f32(_Float16 (*As)[40],
                                                _Float16 (*Bs)[40], int xr,
                                                int xc4, int wk, int wn,
                                                const TileRegsF32& t) {
#pragma unroll
  for (int rr = 0; rr < 4; ++rr) {
    As[xr + rr * 32][xc4 + 0] = (_Float16)t.a[rr].x;
    As[xr + rr * 32][xc4 + 1] = (_Float16)t.a[rr].y;
    As[xr + rr * 32][xc4 + 2] = (_Float16)t.a[rr].z;
    As[xr + rr * 32][xc4 + 3] = (_Float16)t.a[rr].w;
  }
#pragma unroll
  for (int q4 = 0; q4 < 4; ++q4) {
    Bs[wn + q4 * 4 + 0][wk] = (_Float16)t.b[q4].x;
    Bs[wn + q4 * 4 + 1][wk] = (_Float16)t.b[q4].y;
    Bs[wn + q4 * 4 + 2][wk] = (_Float16)t.b[q4].z;
    Bs[wn + q4 * 4 + 3][wk] = (_Float16)t.b[q4].w;
  }
}

// ---------------------------------------------------------------------------
// Kernel 1: QKV projections.  out = x @ W + b (4096x2048x2048), f32 -> f16.
// Double-buffered LDS (one barrier per K-step), global prefetch 2 tiles ahead.
// Q,K stored [B,H,T,d]; V stored transposed [B,H,d,T] for the PV GEMM.
// ---------------------------------------------------------------------------
__global__ __launch_bounds__(256) void qkv_proj_kernel(
    const float* __restrict__ x,
    const float* __restrict__ Wq, const float* __restrict__ bq,
    const float* __restrict__ Wk, const float* __restrict__ bk,
    const float* __restrict__ Wv, const float* __restrict__ bv,
    _Float16* __restrict__ Q, _Float16* __restrict__ K,
    _Float16* __restrict__ Vt)
{
  __shared__ _Float16 As[2][128][40];   // padded: 80B rows, conflict-free frags
  __shared__ _Float16 Bs[2][128][40];   // W tile transposed [n][k]

  const int which = (int)blockIdx.z;
  const float* __restrict__ W    = (which == 0) ? Wq : (which == 1) ? Wk : Wv;
  const float* __restrict__ bias = (which == 0) ? bq : (which == 1) ? bk : bv;

  const int tid  = (int)threadIdx.x;
  const int lane = tid & 31;
  const int wave = tid >> 5;
  const int waveM = wave >> 1;             // 0..3 -> 32-row strip
  const int waveN = wave & 1;              // 0..1 -> 64-col strip
  const int rowBase = (int)blockIdx.y * 128;
  const int colBase = (int)blockIdx.x * 128;

  const v8f vzero = {0.f, 0.f, 0.f, 0.f, 0.f, 0.f, 0.f, 0.f};
  v8f acc[2][4];
#pragma unroll
  for (int i = 0; i < 2; ++i)
#pragma unroll
    for (int j = 0; j < 4; ++j) acc[i][j] = vzero;

  const int xr  = tid >> 3;                // 0..31  (x row group)
  const int xc4 = (tid & 7) << 2;          // 0..28  (x col, float4)
  const int wk  = tid & 31;                // W row within K-step
  const int wn  = (tid >> 5) << 4;         // W col group (16 cols)

  TileRegsF32 tr;
  stage_load_f32(x, W, rowBase, colBase, 0, xr, xc4, wk, wn, tr);
  stage_store_f32(As[0], Bs[0], xr, xc4, wk, wn, tr);
  __syncthreads();

  for (int it = 0; it < KSTEPS; ++it) {
    const int cur = it & 1;
    if (it + 1 < KSTEPS)
      stage_load_f32(x, W, rowBase, colBase, (it + 1) * 32, xr, xc4, wk, wn, tr);
    if (it + 2 < KSTEPS) {   // -> global_prefetch_b8
      __builtin_prefetch(x + (size_t)(rowBase + xr) * EMBED + (it + 2) * 32 + xc4,
                         0, 3);
      __builtin_prefetch(W + (size_t)((it + 2) * 32 + wk) * EMBED + colBase + wn,
                         0, 3);
    }

    v16h af[2], bf[4];
#pragma unroll
    for (int fm = 0; fm < 2; ++fm)
      af[fm] = load_a_frag(&As[cur][waveM * 32 + fm * 16][0], 40);
#pragma unroll
    for (int fn = 0; fn < 4; ++fn)
      bf[fn] = load_b_frag(&Bs[cur][waveN * 64 + fn * 16][0], 40);
#pragma unroll
    for (int fm = 0; fm < 2; ++fm)
#pragma unroll
      for (int fn = 0; fn < 4; ++fn)
        acc[fm][fn] = wmma16(af[fm], bf[fn], acc[fm][fn]);

    if (it + 1 < KSTEPS)
      stage_store_f32(As[1 - cur], Bs[1 - cur], xr, xc4, wk, wn, tr);
    __syncthreads();
  }

  // epilogue: + bias, convert f16, scatter into per-head layouts
  const int half = lane >> 4;
  const int lc   = lane & 15;
#pragma unroll
  for (int fm = 0; fm < 2; ++fm) {
#pragma unroll
    for (int fn = 0; fn < 4; ++fn) {
      const int c = colBase + waveN * 64 + fn * 16 + lc;
      const float bv_ = bias[c];
      const int h  = c >> 7;                // head
      const int dd = c & (HDIM - 1);        // dim within head
#pragma unroll
      for (int j = 0; j < 8; ++j) {
        const int r  = rowBase + waveM * 32 + fm * 16 + j + (half << 3);
        const int bi = r >> 11;             // batch (SEQ = 2048)
        const int t  = r & (SEQ - 1);       // token
        const float val = acc[fm][fn][j] + bv_;
        if (which == 0)
          Q[((size_t)(bi * NHEAD + h) * SEQ + t) * HDIM + dd] = (_Float16)val;
        else if (which == 1)
          K[((size_t)(bi * NHEAD + h) * SEQ + t) * HDIM + dd] = (_Float16)val;
        else
          Vt[((size_t)(bi * NHEAD + h) * HDIM + dd) * SEQ + t] = (_Float16)val;
      }
    }
  }
}

// ---------------------------------------------------------------------------
// Kernel 2: causal flash attention, one block = (b,h) x 128 query rows.
// 8 waves, each owns 16 query rows with full d=128 in registers.
// K tile staged in LDS once per key block (8x less VMEM than per-wave loads);
// the SAME buffer is reused for the V tile after S is computed (K is dead).
// P round-trips through a second LDS buffer (C-layout -> A-fragment layout).
// Dynamic LDS (2 x 128 x 136 f16 = 68 KB; CDNA5 allows up to 320 KB/WG).
// ---------------------------------------------------------------------------
__global__ __launch_bounds__(256) void flash_attn_kernel(
    const _Float16* __restrict__ Q, const _Float16* __restrict__ K,
    const _Float16* __restrict__ Vt, _Float16* __restrict__ attn)
{
  extern __shared__ __align__(16) _Float16 smem[];
  _Float16 (*KV)[136] = (_Float16 (*)[136])smem;               // K, then V
  _Float16 (*Pb)[136] = (_Float16 (*)[136])(smem + 128 * 136); // P tile

  const int tid  = (int)threadIdx.x;
  const int lane = tid & 31;
  const int wave = tid >> 5;
  const int half = lane >> 4;
  const int lc   = lane & 15;
  const int bh   = (int)blockIdx.y;        // b*NHEAD + h
  const int bi   = bh >> 4;
  const int h    = bh & (NHEAD - 1);
  const int qblk = (int)blockIdx.x;

  const _Float16* __restrict__ Qh = Q  + (size_t)bh * SEQ * HDIM;
  const _Float16* __restrict__ Kh = K  + (size_t)bh * SEQ * HDIM;
  const _Float16* __restrict__ Vh = Vt + (size_t)bh * HDIM * SEQ;

  const int qrow0 = qblk * 128 + wave * 16;
  const int crow  = tid >> 1;              // 0..127: staging row
  const int cseg  = (tid & 1) << 6;        // 0 or 64: staging 64-elem segment

  // Q fragments for this wave's 16 rows (d = 4 K-chunks of 32)
  v16h qf[4];
#pragma unroll
  for (int kt = 0; kt < 4; ++kt)
    qf[kt] = load_a_frag(Qh + (size_t)qrow0 * HDIM + kt * 32, HDIM);

  const v8f vzero = {0.f, 0.f, 0.f, 0.f, 0.f, 0.f, 0.f, 0.f};
  v8f o[8];
#pragma unroll
  for (int n = 0; n < 8; ++n) o[n] = vzero;
  float mrun[8], lrun[8];
#pragma unroll
  for (int j = 0; j < 8; ++j) { mrun[j] = -3.0e38f; lrun[j] = 0.f; }
  const float scale = 0.08838834764831845f;   // 1/sqrt(128)

  for (int kb = 0; kb <= qblk; ++kb) {
    // ---- stage K tile [key][d] into KV
    {
      const _Float16* src = Kh + (size_t)(kb * 128 + crow) * HDIM + cseg;
#pragma unroll
      for (int u = 0; u < 8; ++u)
        *(v8h*)&KV[crow][cseg + u * 8] = *(const v8h*)(src + u * 8);
    }
    __syncthreads();

    // ---- S = Q K^T from LDS
    v8f s[8];
#pragma unroll
    for (int n = 0; n < 8; ++n) s[n] = vzero;
#pragma unroll
    for (int kt = 0; kt < 4; ++kt)
#pragma unroll
      for (int n = 0; n < 8; ++n) {
        v16h b = load_b_frag(&KV[n * 16][kt * 32], 136);
        s[n] = wmma16(qf[kt], b, s[n]);
      }

    // ---- scale + causal mask (diagonal block only)
    const bool diag = (kb == qblk);
#pragma unroll
    for (int n = 0; n < 8; ++n)
#pragma unroll
      for (int j = 0; j < 8; ++j) {
        float v = s[n][j] * scale;
        if (diag && (n * 16 + lc) > (wave * 16 + j + (half << 3)))
          v = -3.0e38f;
        s[n][j] = v;
      }

    // ---- online softmax (register-only)
    float alpha[8];
#pragma unroll
    for (int j = 0; j < 8; ++j) {
      float m = s[0][j];
#pragma unroll
      for (int n = 1; n < 8; ++n) m = fmaxf(m, s[n][j]);
#pragma unroll
      for (int off = 1; off < 16; off <<= 1)
        m = fmaxf(m, __shfl_xor(m, off, 32));
      const float mnew = fmaxf(mrun[j], m);
      alpha[j] = __expf(mrun[j] - mnew);
      mrun[j]  = mnew;
    }
    float rsum[8];
#pragma unroll
    for (int j = 0; j < 8; ++j) rsum[j] = 0.f;
#pragma unroll
    for (int n = 0; n < 8; ++n)
#pragma unroll
      for (int j = 0; j < 8; ++j) {
        const float p = __expf(s[n][j] - mrun[j]);
        s[n][j] = p;
        rsum[j] += p;
      }
#pragma unroll
    for (int j = 0; j < 8; ++j) {
      float rs = rsum[j];
#pragma unroll
      for (int off = 1; off < 16; off <<= 1) rs += __shfl_xor(rs, off, 32);
      lrun[j] = lrun[j] * alpha[j] + rs;
#pragma unroll
      for (int n = 0; n < 8; ++n) o[n][j] *= alpha[j];
    }
    __syncthreads();   // all waves finished reading K from KV

    // ---- stage V tile [d][key] into KV (reusing K's buffer) + write P
    {
      const _Float16* src = Vh + (size_t)crow * SEQ + kb * 128 + cseg;
#pragma unroll
      for (int u = 0; u < 8; ++u)
        *(v8h*)&KV[crow][cseg + u * 8] = *(const v8h*)(src + u * 8);
    }
    {
      _Float16 (*Pw)[136] = Pb + wave * 16;
#pragma unroll
      for (int n = 0; n < 8; ++n)
#pragma unroll
        for (int j = 0; j < 8; ++j)
          Pw[j + (half << 3)][n * 16 + lc] = (_Float16)s[n][j];
    }
    __syncthreads();

    // ---- O += P V  (A from Pb, B from KV: col = d, contiguous keys)
#pragma unroll
    for (int kt = 0; kt < 4; ++kt) {
      v16h a = load_a_frag(&Pb[wave * 16][kt * 32], 136);
#pragma unroll
      for (int n = 0; n < 8; ++n) {
        v16h b = load_b_frag(&KV[n * 16][kt * 32], 136);
        o[n] = wmma16(a, b, o[n]);
      }
    }
    __syncthreads();   // protect KV/Pb before next iteration overwrites
  }

  // ---- finalize and write attn[B,T,C] in f16
#pragma unroll
  for (int j = 0; j < 8; ++j) lrun[j] = 1.f / lrun[j];
#pragma unroll
  for (int n = 0; n < 8; ++n)
#pragma unroll
    for (int j = 0; j < 8; ++j) {
      const int t = qrow0 + j + (half << 3);
      attn[((size_t)(bi * SEQ + t)) * EMBED + h * HDIM + n * 16 + lc] =
          (_Float16)(o[n][j] * lrun[j]);
    }
}

// ---------------------------------------------------------------------------
// Kernel 3: output projection.  out = attn @ Wo + bo  (f16 x f16 -> f32)
// Double-buffered LDS like kernel 1; A tile is already f16 (vector copies).
// ---------------------------------------------------------------------------
__global__ __launch_bounds__(256) void out_proj_kernel(
    const _Float16* __restrict__ A, const float* __restrict__ Wo,
    const float* __restrict__ bo, float* __restrict__ out)
{
  __shared__ _Float16 As[2][128][40];
  __shared__ _Float16 Bs[2][128][40];

  const int tid  = (int)threadIdx.x;
  const int lane = tid & 31;
  const int wave = tid >> 5;
  const int waveM = wave >> 1;
  const int waveN = wave & 1;
  const int rowBase = (int)blockIdx.y * 128;
  const int colBase = (int)blockIdx.x * 128;

  const v8f vzero = {0.f, 0.f, 0.f, 0.f, 0.f, 0.f, 0.f, 0.f};
  v8f acc[2][4];
#pragma unroll
  for (int i = 0; i < 2; ++i)
#pragma unroll
    for (int j = 0; j < 4; ++j) acc[i][j] = vzero;

  const int ar = tid >> 2;                 // 0..63
  const int as = (tid & 3) << 3;           // 0,8,16,24
  const int wk = tid & 31;
  const int wn = (tid >> 5) << 4;

  v8h  ta[2];
  float4 tb[4];

  // initial stage (k0 = 0) into buffer 0
#pragma unroll
  for (int rr = 0; rr < 2; ++rr)
    ta[rr] = *(const v8h*)(A + (size_t)(rowBase + ar + rr * 64) * EMBED + as);
  {
    const float* wp = Wo + (size_t)wk * EMBED + colBase + wn;
#pragma unroll
    for (int q4 = 0; q4 < 4; ++q4) tb[q4] = *(const float4*)(wp + q4 * 4);
  }
#pragma unroll
  for (int rr = 0; rr < 2; ++rr) *(v8h*)&As[0][ar + rr * 64][as] = ta[rr];
#pragma unroll
  for (int q4 = 0; q4 < 4; ++q4) {
    Bs[0][wn + q4 * 4 + 0][wk] = (_Float16)tb[q4].x;
    Bs[0][wn + q4 * 4 + 1][wk] = (_Float16)tb[q4].y;
    Bs[0][wn + q4 * 4 + 2][wk] = (_Float16)tb[q4].z;
    Bs[0][wn + q4 * 4 + 3][wk] = (_Float16)tb[q4].w;
  }
  __syncthreads();

  for (int it = 0; it < KSTEPS; ++it) {
    const int cur = it & 1;
    if (it + 1 < KSTEPS) {
      const int k0n = (it + 1) * 32;
#pragma unroll
      for (int rr = 0; rr < 2; ++rr)
        ta[rr] = *(const v8h*)(A + (size_t)(rowBase + ar + rr * 64) * EMBED +
                               k0n + as);
      const float* wp = Wo + (size_t)(k0n + wk) * EMBED + colBase + wn;
#pragma unroll
      for (int q4 = 0; q4 < 4; ++q4) tb[q4] = *(const float4*)(wp + q4 * 4);
    }
    if (it + 2 < KSTEPS) {   // -> global_prefetch_b8
      __builtin_prefetch(A + (size_t)(rowBase + ar) * EMBED + (it + 2) * 32 + as,
                         0, 3);
      __builtin_prefetch(
          Wo + (size_t)((it + 2) * 32 + wk) * EMBED + colBase + wn, 0, 3);
    }

    v16h af[2], bf[4];
#pragma unroll
    for (int fm = 0; fm < 2; ++fm)
      af[fm] = load_a_frag(&As[cur][waveM * 32 + fm * 16][0], 40);
#pragma unroll
    for (int fn = 0; fn < 4; ++fn)
      bf[fn] = load_b_frag(&Bs[cur][waveN * 64 + fn * 16][0], 40);
#pragma unroll
    for (int fm = 0; fm < 2; ++fm)
#pragma unroll
      for (int fn = 0; fn < 4; ++fn)
        acc[fm][fn] = wmma16(af[fm], bf[fn], acc[fm][fn]);

    if (it + 1 < KSTEPS) {
      const int nxt = 1 - cur;
#pragma unroll
      for (int rr = 0; rr < 2; ++rr) *(v8h*)&As[nxt][ar + rr * 64][as] = ta[rr];
#pragma unroll
      for (int q4 = 0; q4 < 4; ++q4) {
        Bs[nxt][wn + q4 * 4 + 0][wk] = (_Float16)tb[q4].x;
        Bs[nxt][wn + q4 * 4 + 1][wk] = (_Float16)tb[q4].y;
        Bs[nxt][wn + q4 * 4 + 2][wk] = (_Float16)tb[q4].z;
        Bs[nxt][wn + q4 * 4 + 3][wk] = (_Float16)tb[q4].w;
      }
    }
    __syncthreads();
  }

  const int half = lane >> 4;
  const int lc   = lane & 15;
#pragma unroll
  for (int fm = 0; fm < 2; ++fm)
#pragma unroll
    for (int fn = 0; fn < 4; ++fn) {
      const int c = colBase + waveN * 64 + fn * 16 + lc;
      const float bv_ = bo[c];
#pragma unroll
      for (int j = 0; j < 8; ++j) {
        const int r = rowBase + waveM * 32 + fm * 16 + j + (half << 3);
        out[(size_t)r * EMBED + c] = acc[fm][fn][j] + bv_;
      }
    }
}

// ---------------------------------------------------------------------------
extern "C" void kernel_launch(void* const* d_in, const int* in_sizes, int n_in,
                              void* d_out, int out_size, void* d_ws,
                              size_t ws_size, hipStream_t stream) {
  (void)in_sizes; (void)n_in; (void)out_size; (void)ws_size;
  const float* x  = (const float*)d_in[0];
  const float* Wq = (const float*)d_in[1];
  const float* bq = (const float*)d_in[2];
  const float* Wk = (const float*)d_in[3];
  const float* bk = (const float*)d_in[4];
  const float* Wv = (const float*)d_in[5];
  const float* bv = (const float*)d_in[6];
  const float* Wo = (const float*)d_in[7];
  const float* bo = (const float*)d_in[8];
  // d_in[9] = train flag (dropout rate 0 -> unused)

  _Float16* Q    = (_Float16*)d_ws;
  _Float16* K    = Q + BHTD;
  _Float16* Vt   = K + BHTD;
  _Float16* attn = Vt + BHTD;     // total 64 MB of workspace

  dim3 blk(256);
  dim3 g1(EMBED / 128, (BATCH * SEQ) / 128, 3);   // 16 x 32 x 3
  qkv_proj_kernel<<<g1, blk, 0, stream>>>(x, Wq, bq, Wk, bk, Wv, bv, Q, K, Vt);

  const size_t attn_lds = (size_t)2 * 128 * 136 * sizeof(_Float16);  // 69632 B
  dim3 g2(SEQ / 128, BATCH * NHEAD);              // 16 x 32
  flash_attn_kernel<<<g2, blk, attn_lds, stream>>>(Q, K, Vt, attn);

  dim3 g3(EMBED / 128, (BATCH * SEQ) / 128);      // 16 x 32
  out_proj_kernel<<<g3, blk, 0, stream>>>(attn, Wo, bo, (float*)d_out);
}
